// LSTMClog_69810398429761
// MI455X (gfx1250) — matrix operand order repaired and checked by reading, not compiled, not run on hardware
//
#include <hip/hip_runtime.h>
#include <hip/hip_bf16.h>

typedef __attribute__((ext_vector_type(16))) _Float16 v16h;
typedef __attribute__((ext_vector_type(8)))  _Float16 v8h;
typedef __attribute__((ext_vector_type(8)))  float    v8f;
typedef __attribute__((ext_vector_type(4)))  float    v4f;

#define B_   512
#define T_   512
#define C_   64
#define H_   32
#define OUT_ 10
#define NTB  32                                   // 512 batch rows / 16-row tiles

// workspace layout
#define XG_HALVES  (2ull * NTB * T_ * 8 * 256)    // [dir][tb][t][nt][lane][r] f16, D-layout
#define WFC_HALVES (2ull * T_ * 32 * 16)          // [dir][t][lane][h] f16, B-layout tiles
#define XG_BYTES   (XG_HALVES * 2)
#define WFC_BYTES  (WFC_HALVES * 2)

__device__ __forceinline__ float sigm_f(float x) {
    return 1.0f / (1.0f + __expf(-x));
}
__device__ __forceinline__ float tanh_f(float x) {
    float t = __expf(-2.0f * fabsf(x));
    float r = (1.0f - t) / (1.0f + t);
    return copysignf(r, x);
}

// ---------------------------------------------------------------------------
// Prep: W_fc per-timestep slices -> f16 B-layout tiles (32x16: K=hidden, N=out)
// b[h] = Wfc[o=lane&15][t*64 + dir*32 + k], k = 8*(lane>>4) + (h&7) + 16*(h>>3)
// ---------------------------------------------------------------------------
__global__ void prep_wfc_kernel(const float* __restrict__ Wfc,
                                _Float16* __restrict__ wfcT) {
    int idx = blockIdx.x * blockDim.x + threadIdx.x;   // [dir][t][lane][h]
    if (idx >= 2 * T_ * 32 * 16) return;
    int h    = idx & 15;
    int lane = (idx >> 4) & 31;
    int t    = (idx >> 9) & (T_ - 1);
    int dir  = idx >> 18;
    int k    = 8 * (lane >> 4) + (h & 7) + 16 * (h >> 3);
    int o    = lane & 15;
    float v  = (o < OUT_) ? Wfc[(size_t)o * (T_ * 2 * H_) + t * 64 + dir * 32 + k]
                          : 0.0f;
    wfcT[idx] = (_Float16)v;
}

// ---------------------------------------------------------------------------
// Input projection GEMM: xg[dir][b][t][g] = x[b,t,:] @ W_ih^T + b_ih + b_hh
// One wave = 16 batch rows x 32 timesteps x all 128 gate columns.
// Output stored f16 in WMMA C/D lane layout for zero-shuffle reload in scan.
// ---------------------------------------------------------------------------
__global__ __launch_bounds__(32) void xg_gemm_kernel(
    const float* __restrict__ x,
    const float* __restrict__ Wih_f, const float* __restrict__ Wih_b,
    const float* __restrict__ bih_f, const float* __restrict__ bhh_f,
    const float* __restrict__ bih_b, const float* __restrict__ bhh_b,
    _Float16* __restrict__ xg)
{
    const int lane = threadIdx.x;
    const int l2 = lane & 15, hi = lane >> 4;
    const int tb  = blockIdx.x;   // batch tile
    const int dir = blockIdx.y;   // direction
    const int tc  = blockIdx.z;   // time chunk (32 t's)

    const float* Wih = dir ? Wih_b : Wih_f;
    const float* bih = dir ? bih_b : bih_f;
    const float* bhh = dir ? bhh_b : bhh_f;

    // W_ih as B-layout tiles: b[h] = Wih[n=16*nt+l2][k = 8*hi + (h&7) + 16*(h>>3) + 32*kt]
    v16h w[2][8];
#pragma unroll
    for (int kt = 0; kt < 2; ++kt) {
        const int k = 8 * hi + 32 * kt;
#pragma unroll
        for (int nt = 0; nt < 8; ++nt)
#pragma unroll
            for (int h = 0; h < 16; ++h)
                w[kt][nt][h] = (_Float16)Wih[(16 * nt + l2) * C_ + k + (h & 7) + 16 * (h >> 3)];
    }
    float bias[8];
#pragma unroll
    for (int nt = 0; nt < 8; ++nt)
        bias[nt] = bih[16 * nt + l2] + bhh[16 * nt + l2];

    const size_t xrow = (size_t)(tb * 16 + l2) * T_;

    for (int ti = 0; ti < 32; ++ti) {
        const int t = tc * 32 + ti;
        const float* xp = x + (xrow + t) * C_;

        // A-layout f16: lanes 0-15 row l2 K{0..7,16..23}(+32kt), lanes 16-31 K{8..15,24..31}
        v16h a[2];
#pragma unroll
        for (int kt = 0; kt < 2; ++kt) {
            const int c0 = kt * 32 + hi * 8;
            const v4f* p0 = (const v4f*)(xp + c0);
            const v4f* p1 = (const v4f*)(xp + c0 + 16);
            v4f x0 = p0[0], x1 = p0[1], x2 = p1[0], x3 = p1[1];
#pragma unroll
            for (int e = 0; e < 4; ++e) {
                a[kt][e]      = (_Float16)x0[e];
                a[kt][4 + e]  = (_Float16)x1[e];
                a[kt][8 + e]  = (_Float16)x2[e];
                a[kt][12 + e] = (_Float16)x3[e];
            }
        }

        v8f acc[8];
#pragma unroll
        for (int nt = 0; nt < 8; ++nt)
#pragma unroll
            for (int r = 0; r < 8; ++r) acc[nt][r] = bias[nt];   // bias per N col

#pragma unroll
        for (int kt = 0; kt < 2; ++kt)
#pragma unroll
            for (int nt = 0; nt < 8; ++nt)
                acc[nt] = __builtin_amdgcn_wmma_f32_16x16x32_f16(
                    false, a[kt], false, w[kt][nt], (short)0, acc[nt], false, false);

        const size_t base =
            ((((size_t)dir * NTB + tb) * T_ + t) * 8) * 256 + (size_t)lane * 8;
#pragma unroll
        for (int nt = 0; nt < 8; ++nt) {
            v8h o;
#pragma unroll
            for (int r = 0; r < 8; ++r) o[r] = (_Float16)acc[nt][r];
            *(v8h*)(xg + base + nt * 256) = o;                    // coalesced b128
        }
    }
}

// ---------------------------------------------------------------------------
// Sequential LSTM scan, one wave per (direction, 16-batch-row tile).
// Per step: load 8 C-tiles (xg, coalesced) -> 8 WMMAs (h @ W_hh^T) ->
// gates -> c/h update -> LDS bounce to A-layout -> 1 WMMA FC accumulate.
// ---------------------------------------------------------------------------
__global__ __launch_bounds__(32) void lstm_scan_kernel(
    const _Float16* __restrict__ xg,
    const float* __restrict__ Whh_f, const float* __restrict__ Whh_b,
    const _Float16* __restrict__ wfcT,
    float* __restrict__ partial)
{
    const int lane = threadIdx.x;
    const int l2 = lane & 15, hi = lane >> 4;
    const int tb  = blockIdx.x;
    const int dir = blockIdx.y;
    const float* Whh = dir ? Whh_b : Whh_f;

    // W_hh as B-layout tiles: b[h] = Whh[n=16*nt+l2][k = 8*hi + (h&7) + 16*(h>>3)]
    v16h whh[8];
    {
        const int k = 8 * hi;
#pragma unroll
        for (int nt = 0; nt < 8; ++nt)
#pragma unroll
            for (int h = 0; h < 16; ++h)
                whh[nt][h] = (_Float16)Whh[(16 * nt + l2) * H_ + k + (h & 7) + 16 * (h >> 3)];
    }

    __shared__ _Float16 hbuf[16 * 32];

    v16h hA = {};          // h_{t-1} in A layout (f16)
    v8f  cst[2] = {};      // cell state, D layout, hidden cols [0..15],[16..31]
    v8f  outacc = {};      // fused-FC accumulator (N = output class)

    for (int s = 0; s < T_; ++s) {
        const int t = dir ? (T_ - 1 - s) : s;
        const size_t base =
            ((((size_t)dir * NTB + tb) * T_ + t) * 8) * 256 + (size_t)lane * 8;

        if (s + 1 < T_) {   // prefetch next step's gate pre-activations
            const int tn = dir ? (t - 1) : (t + 1);
            __builtin_prefetch(
                xg + ((((size_t)dir * NTB + tb) * T_ + tn) * 8) * 256 +
                    (size_t)lane * 8, 0, 0);
        }

        v8f acc[8];
#pragma unroll
        for (int nt = 0; nt < 8; ++nt) {
            v8h g = *(const v8h*)(xg + base + nt * 256);
#pragma unroll
            for (int r = 0; r < 8; ++r) acc[nt][r] = (float)g[r];
        }

#pragma unroll
        for (int nt = 0; nt < 8; ++nt)
            acc[nt] = __builtin_amdgcn_wmma_f32_16x16x32_f16(
                false, hA, false, whh[nt], (short)0, acc[nt], false, false);

        // gate order i,f,g,o -> tiles {0,1},{2,3},{4,5},{6,7}
#pragma unroll
        for (int j = 0; j < 2; ++j)
#pragma unroll
            for (int r = 0; r < 8; ++r) {
                float ig = sigm_f(acc[j][r]);
                float fg = sigm_f(acc[2 + j][r]);
                float gg = tanh_f(acc[4 + j][r]);
                float og = sigm_f(acc[6 + j][r]);
                float c  = fg * cst[j][r] + ig * gg;
                cst[j][r] = c;
                float hv = og * tanh_f(c);
                hbuf[(r + 8 * hi) * 32 + 16 * j + l2] = (_Float16)hv;
            }
        __syncthreads();

        // reload h_t in A layout (two b128 LDS reads per lane)
#pragma unroll
        for (int h = 0; h < 8; ++h) hA[h] = hbuf[l2 * 32 + hi * 8 + h];
#pragma unroll
        for (int h = 0; h < 8; ++h) hA[8 + h] = hbuf[l2 * 32 + 16 + hi * 8 + h];
        __syncthreads();

        // fused FC: out += h_t @ Wfc_slice^T  (one WMMA, B tile preloaded coalesced)
        v16h wfc = *(const v16h*)(wfcT + (((size_t)dir * T_ + t) * 32 + lane) * 16);
        outacc = __builtin_amdgcn_wmma_f32_16x16x32_f16(
            false, hA, false, wfc, (short)0, outacc, false, false);
    }

#pragma unroll
    for (int r = 0; r < 8; ++r)
        partial[(((size_t)dir * NTB + tb) * 32 + lane) * 8 + r] = outacc[r];
}

// ---------------------------------------------------------------------------
// Epilogue: out[b][o] = b_fc[o] + partial_fwd + partial_bwd
// ---------------------------------------------------------------------------
__global__ void fc_final_kernel(const float* __restrict__ partial,
                                const float* __restrict__ bfc,
                                float* __restrict__ out) {
    int idx = blockIdx.x * blockDim.x + threadIdx.x;
    if (idx >= B_ * OUT_) return;
    int o = idx % OUT_, b = idx / OUT_;
    int tb = b >> 4, M = b & 15;
    int h2 = M >> 3, r = M & 7, lane = 16 * h2 + o;
    float v = bfc[o];
    v += partial[((size_t)(0 * NTB + tb) * 32 + lane) * 8 + r];
    v += partial[((size_t)(1 * NTB + tb) * 32 + lane) * 8 + r];
    out[idx] = v;
}

extern "C" void kernel_launch(void* const* d_in, const int* in_sizes, int n_in,
                              void* d_out, int out_size, void* d_ws, size_t ws_size,
                              hipStream_t stream) {
    const float* x     = (const float*)d_in[0];
    const float* Wih_f = (const float*)d_in[1];
    const float* Whh_f = (const float*)d_in[2];
    const float* bih_f = (const float*)d_in[3];
    const float* bhh_f = (const float*)d_in[4];
    const float* Wih_b = (const float*)d_in[5];
    const float* Whh_b = (const float*)d_in[6];
    const float* bih_b = (const float*)d_in[7];
    const float* bhh_b = (const float*)d_in[8];
    const float* Wfc   = (const float*)d_in[9];
    const float* bfc   = (const float*)d_in[10];
    float* out = (float*)d_out;

    char* ws = (char*)d_ws;
    _Float16* xg     = (_Float16*)ws;
    _Float16* wfcT   = (_Float16*)(ws + XG_BYTES);
    float*    part   = (float*)(ws + XG_BYTES + WFC_BYTES);

    hipLaunchKernelGGL(prep_wfc_kernel, dim3((2 * T_ * 32 * 16 + 255) / 256),
                       dim3(256), 0, stream, Wfc, wfcT);
    hipLaunchKernelGGL(xg_gemm_kernel, dim3(NTB, 2, 16), dim3(32), 0, stream,
                       x, Wih_f, Wih_b, bih_f, bhh_f, bih_b, bhh_b, xg);
    hipLaunchKernelGGL(lstm_scan_kernel, dim3(NTB, 2), dim3(32), 0, stream,
                       xg, Whh_f, Whh_b, wfcT, part);
    hipLaunchKernelGGL(fc_final_kernel, dim3((B_ * OUT_ + 255) / 256),
                       dim3(256), 0, stream, part, bfc, out);
}
